// PointTransformerLayer_7808250544220
// MI455X (gfx1250) — matrix operand order repaired and checked by reading, not compiled, not run on hardware
//
#include <hip/hip_runtime.h>
#include <hip/hip_bf16.h>

typedef _Float16 half_t;
typedef __attribute__((ext_vector_type(16))) _Float16 v16h;
typedef __attribute__((ext_vector_type(8)))  _Float16 v8h;
typedef __attribute__((ext_vector_type(8)))  float    v8f;

#define NPTS  1024
#define DD    64
#define HPOS  64
#define HATTN 256

// D = A(16x32 f16) * B(32x16 f16) + C(16x16 f32), wave32
__device__ __forceinline__ v8f wmma_f16(v16h a, v16h b, v8f c) {
    return __builtin_amdgcn_wmma_f32_16x16x32_f16(
        /*neg_a=*/false, a, /*neg_b=*/false, b,
        /*c_mod=*/(short)0, c, /*reuse_a=*/false, /*reuse_b=*/false);
}

// A-fragment (16x32, f16) from LDS tile row-major [16][K].
// p points at &tile[m][kt*32]; lane layout: m = lane&15, hi = lane>>4.
// a[e] = K(hi*16 + e)
__device__ __forceinline__ v16h load_A(const half_t* p, int hi) {
    v8h lo = *reinterpret_cast<const v8h*>(p + hi * 16);
    v8h hv = *reinterpret_cast<const v8h*>(p + hi * 16 + 8);
    v16h a;
#pragma unroll
    for (int e = 0; e < 8; ++e) { a[e] = lo[e]; a[e + 8] = hv[e]; }
    return a;
}

// B-fragment (32x16, f16) from TRANSPOSED LDS weights Wt[n][k].
// p points at &Wt[nt*16 + (lane&15)][kt*32 + hi*16]; b[e] = K(hi*16 + e)
__device__ __forceinline__ v16h load_B(const half_t* p) {
    v8h b0 = *reinterpret_cast<const v8h*>(p);
    v8h b1 = *reinterpret_cast<const v8h*>(p + 8);
    v16h b;
#pragma unroll
    for (int e = 0; e < 8; ++e) { b[e] = b0[e]; b[e + 8] = b1[e]; }
    return b;
}

// ---------------------------------------------------------------------------
// Kernel 1: q/k/v = x @ {Wq,Wk,Wv}.  grid=(64 row-tiles, 3), block=128 (4 waves).
// Wave w computes output columns [w*16, w*16+16).
// ---------------------------------------------------------------------------
__global__ __launch_bounds__(128) void qkv_kernel(
    const float* __restrict__ x,
    const float* __restrict__ Wq, const float* __restrict__ Wk,
    const float* __restrict__ Wv,
    float* __restrict__ q, float* __restrict__ k, float* __restrict__ v)
{
    __shared__ __align__(16) half_t s_x[16][DD];   // A source: [row][k]
    __shared__ __align__(16) half_t s_W[DD][DD];   // B source transposed: [n][k]

    const float* W = (blockIdx.y == 0) ? Wq : (blockIdx.y == 1) ? Wk : Wv;
    float* out     = (blockIdx.y == 0) ? q  : (blockIdx.y == 1) ? k  : v;

    const int tid  = threadIdx.x;
    const int row0 = blockIdx.x * 16;

    for (int idx = tid; idx < 16 * DD; idx += 128)
        s_x[idx / DD][idx % DD] = (half_t)x[(row0 + idx / DD) * DD + idx % DD];
    for (int idx = tid; idx < DD * DD; idx += 128) {
        int kk = idx / DD, nn = idx % DD;
        s_W[nn][kk] = (half_t)W[idx];
    }
    __syncthreads();

    const int wave = tid >> 5, lane = tid & 31;
    const int n16 = lane & 15, hi = lane >> 4;
    const int nt = wave;

    v8f c = {0.f, 0.f, 0.f, 0.f, 0.f, 0.f, 0.f, 0.f};
#pragma unroll
    for (int kt = 0; kt < 2; ++kt) {
        v16h a = load_A(&s_x[n16][kt * 32], hi);
        v16h b = load_B(&s_W[nt * 16 + n16][kt * 32 + hi * 16]);
        c = wmma_f16(a, b, c);
    }
#pragma unroll
    for (int r = 0; r < 8; ++r)
        out[(row0 + r + hi * 8) * DD + nt * 16 + n16] = c[r];
}

// ---------------------------------------------------------------------------
// Kernel 2: fused rpe-MLP + attn-MLP + per-channel online softmax + aggregate.
// grid=256, block=128 (4 waves). Wave w of block b owns query i = b*4+w.
// Loops over 64 j-tiles of 16 keys; everything stays in LDS/registers.
// ---------------------------------------------------------------------------
__global__ __launch_bounds__(128) void pt_layer_kernel(
    const float* __restrict__ q, const float* __restrict__ k,
    const float* __restrict__ v, const float* __restrict__ pos,
    const float* __restrict__ pW1, const float* __restrict__ pb1,
    const float* __restrict__ pW2, const float* __restrict__ pb2,
    const float* __restrict__ aW1, const float* __restrict__ ab1,
    const float* __restrict__ aW2, const float* __restrict__ ab2,
    float* __restrict__ out)
{
    // Weights, transposed for B-fragment loads (f16)
    __shared__ __align__(16) half_t s_aW1[HATTN][DD];   // 32 KB  [n][k]
    __shared__ __align__(16) half_t s_aW2[DD][HATTN];   // 32 KB  [n][k]
    __shared__ __align__(16) half_t s_pW2[DD][HPOS];    //  8 KB  [n][k]
    __shared__ float s_pW1[2 * HPOS];
    __shared__ float s_pb1[HPOS];
    __shared__ float s_ab1[HATTN];
    // Per-wave tiles
    __shared__ __align__(16) half_t s_t1[4][16][HPOS];  //  8 KB pos-MLP hidden
    __shared__ __align__(16) half_t s_h [4][16][DD];    //  8 KB attn-MLP input
    __shared__ __align__(16) half_t s_act[4][16][HATTN];// 32 KB attn-MLP hidden

    const int tid = threadIdx.x;

    for (int idx = tid; idx < DD * HATTN; idx += 128) {     // aW1[k][n], k<64
        int kk = idx / HATTN, nn = idx % HATTN;
        s_aW1[nn][kk] = (half_t)aW1[idx];
    }
    for (int idx = tid; idx < HATTN * DD; idx += 128) {     // aW2[k][n], k<256
        int kk = idx / DD, nn = idx % DD;
        s_aW2[nn][kk] = (half_t)aW2[idx];
    }
    for (int idx = tid; idx < HPOS * DD; idx += 128) {      // pW2[k][n]
        int kk = idx / DD, nn = idx % DD;
        s_pW2[nn][kk] = (half_t)pW2[idx];
    }
    if (tid < 2 * HPOS) s_pW1[tid] = pW1[tid];
    if (tid < HPOS)     s_pb1[tid] = pb1[tid];
    for (int idx = tid; idx < HATTN; idx += 128) s_ab1[idx] = ab1[idx];
    __syncthreads();

    const int wave = tid >> 5, lane = tid & 31;
    const int n16 = lane & 15, hi = lane >> 4;
    const int i = blockIdx.x * 4 + wave;                    // query index

    const float px = pos[i * 2 + 0], py = pos[i * 2 + 1];
    float qv[4], pb2v[4], ab2v[4];
#pragma unroll
    for (int t = 0; t < 4; ++t) {
        qv[t]   = q[i * DD + t * 16 + n16];
        pb2v[t] = pb2[t * 16 + n16];
        ab2v[t] = ab2[t * 16 + n16];
    }

    // online-softmax state: channel d = t*16 + n16 (mirrored in lane^16)
    float m_s[4], l_s[4], acc_s[4];
#pragma unroll
    for (int t = 0; t < 4; ++t) { m_s[t] = -1e30f; l_s[t] = 0.f; acc_s[t] = 0.f; }

    const int h0 = lane * 2;   // pos-MLP hidden channels owned by this lane

    for (int jt = 0; jt < NPTS / 16; ++jt) {
        const int j0 = jt * 16;

        // --- (A) pos-MLP layer 1 (K=2, VALU) -> s_t1 (f16) ---------------
#pragma unroll 4
        for (int j = 0; j < 16; ++j) {
            float rx = px - pos[(j0 + j) * 2 + 0];
            float ry = py - pos[(j0 + j) * 2 + 1];
            float a0 = rx * s_pW1[h0]     + ry * s_pW1[HPOS + h0]     + s_pb1[h0];
            float a1 = rx * s_pW1[h0 + 1] + ry * s_pW1[HPOS + h0 + 1] + s_pb1[h0 + 1];
            s_t1[wave][j][h0]     = (half_t)fmaxf(a0, 0.f);
            s_t1[wave][j][h0 + 1] = (half_t)fmaxf(a1, 0.f);
        }
        __syncthreads();

        // --- (B) rpe = relu(t1) @ pW2 + pb2  (8 WMMAs) --------------------
        v8f rpe_c[4];
#pragma unroll
        for (int t = 0; t < 4; ++t) {
            v8f c = {0.f, 0.f, 0.f, 0.f, 0.f, 0.f, 0.f, 0.f};
#pragma unroll
            for (int kt = 0; kt < 2; ++kt) {
                v16h a = load_A(&s_t1[wave][n16][kt * 32], hi);
                v16h b = load_B(&s_pW2[t * 16 + n16][kt * 32 + hi * 16]);
                c = wmma_f16(a, b, c);
            }
#pragma unroll
            for (int r = 0; r < 8; ++r) c[r] += pb2v[t];
            rpe_c[t] = c;
        }

        // --- (C) h = q - k + rpe -> s_h (f16);  vv = v + rpe (regs) -------
        v8f vv[4];
#pragma unroll
        for (int t = 0; t < 4; ++t) {
            const int d = t * 16 + n16;
#pragma unroll
            for (int r = 0; r < 8; ++r) {
                const int j = r + hi * 8;
                float rp = rpe_c[t][r];
                s_h[wave][j][d] = (half_t)(qv[t] - k[(j0 + j) * DD + d] + rp);
                vv[t][r] = v[(j0 + j) * DD + d] + rp;
            }
        }
        __syncthreads();

        // --- (D) layer 1: relu(h @ aW1 + ab1) -> s_act  (32 WMMAs) --------
#pragma unroll
        for (int nt = 0; nt < 16; ++nt) {
            v8f c = {0.f, 0.f, 0.f, 0.f, 0.f, 0.f, 0.f, 0.f};
#pragma unroll
            for (int kt = 0; kt < 2; ++kt) {
                v16h a = load_A(&s_h[wave][n16][kt * 32], hi);
                v16h b = load_B(&s_aW1[nt * 16 + n16][kt * 32 + hi * 16]);
                c = wmma_f16(a, b, c);
            }
            const float bias = s_ab1[nt * 16 + n16];
#pragma unroll
            for (int r = 0; r < 8; ++r)
                s_act[wave][r + hi * 8][nt * 16 + n16] = (half_t)fmaxf(c[r] + bias, 0.f);
        }
        __syncthreads();

        // --- (E) layer 2: sim = act @ aW2 + ab2  (32 WMMAs) ---------------
        v8f sim[4];
#pragma unroll
        for (int t = 0; t < 4; ++t) {
            v8f c = {0.f, 0.f, 0.f, 0.f, 0.f, 0.f, 0.f, 0.f};
#pragma unroll
            for (int kt = 0; kt < 8; ++kt) {
                v16h a = load_A(&s_act[wave][n16][kt * 32], hi);
                v16h b = load_B(&s_aW2[t * 16 + n16][kt * 32 + hi * 16]);
                c = wmma_f16(a, b, c);
            }
#pragma unroll
            for (int r = 0; r < 8; ++r) c[r] += ab2v[t];
            sim[t] = c;
        }

        // --- (F) per-channel online softmax over j ------------------------
        // channel d's 16 rows live in this lane (8 rows) + lane^16 (8 rows)
#pragma unroll
        for (int t = 0; t < 4; ++t) {
            float tmax = -1e30f;
#pragma unroll
            for (int r = 0; r < 8; ++r) tmax = fmaxf(tmax, sim[t][r]);
            tmax = fmaxf(tmax, __shfl_xor(tmax, 16, 32));
            float mnew  = fmaxf(m_s[t], tmax);
            float scale = __expf(m_s[t] - mnew);
            float lsum = 0.f, asum = 0.f;
#pragma unroll
            for (int r = 0; r < 8; ++r) {
                float e = __expf(sim[t][r] - mnew);
                lsum += e;
                asum += e * vv[t][r];
            }
            lsum += __shfl_xor(lsum, 16, 32);
            asum += __shfl_xor(asum, 16, 32);
            l_s[t]   = l_s[t] * scale + lsum;
            acc_s[t] = acc_s[t] * scale + asum;
            m_s[t]   = mnew;
        }
        __syncthreads();   // protect s_t1/s_h/s_act before next iteration
    }

    if (hi == 0) {
#pragma unroll
        for (int t = 0; t < 4; ++t)
            out[i * DD + t * 16 + n16] = acc_s[t] / l_s[t];
    }
}

// ---------------------------------------------------------------------------
extern "C" void kernel_launch(void* const* d_in, const int* in_sizes, int n_in,
                              void* d_out, int out_size, void* d_ws, size_t ws_size,
                              hipStream_t stream) {
    const float* x   = (const float*)d_in[0];
    const float* pos = (const float*)d_in[1];
    const float* Wq  = (const float*)d_in[2];
    const float* Wk  = (const float*)d_in[3];
    const float* Wv  = (const float*)d_in[4];
    const float* pW1 = (const float*)d_in[5];
    const float* pb1 = (const float*)d_in[6];
    const float* pW2 = (const float*)d_in[7];
    const float* pb2 = (const float*)d_in[8];
    const float* aW1 = (const float*)d_in[9];
    const float* ab1 = (const float*)d_in[10];
    const float* aW2 = (const float*)d_in[11];
    const float* ab2 = (const float*)d_in[12];

    float* ws = (float*)d_ws;
    float* qb = ws;
    float* kb = ws + NPTS * DD;
    float* vb = ws + 2 * NPTS * DD;

    qkv_kernel<<<dim3(NPTS / 16, 3), 128, 0, stream>>>(x, Wq, Wk, Wv, qb, kb, vb);
    pt_layer_kernel<<<NPTS / 4, 128, 0, stream>>>(
        qb, kb, vb, pos, pW1, pb1, pW2, pb2, aW1, ab1, aW2, ab2, (float*)d_out);
}
